// GINEEncoder_85933705658440
// MI455X (gfx1250) — matrix operand order repaired and checked by reading, not compiled, not run on hardware
//
#include <hip/hip_runtime.h>

#define NN 50000
#define NE 800000
#define NG 512
#define BN_EPS 1e-5f

typedef __attribute__((ext_vector_type(16))) _Float16 v16h;
typedef __attribute__((ext_vector_type(8)))  _Float16 v8h;
typedef __attribute__((ext_vector_type(8)))  float    v8f;

// ---------------- weight conversion (f32 -> f16) ----------------
__global__ __launch_bounds__(256) void cvt_f16_k(const float* __restrict__ s,
                                                 _Float16* __restrict__ d, int n) {
  int i = blockIdx.x * 256 + threadIdx.x;
  if (i < n) d[i] = (_Float16)s[i];
}

// ---------------- column stats (for bn0 only; conv BN stats are fused) -------
template <int D>
__global__ __launch_bounds__(256) void col_stats_k(const float* __restrict__ in,
                                                   float* __restrict__ acc, int n,
                                                   int rows_per_block) {
  const int tpc = 256 / D;  // threads per column
  int col = threadIdx.x % D;
  int sub = threadIdx.x / D;
  int r0 = blockIdx.x * rows_per_block;
  int r1 = r0 + rows_per_block; if (r1 > n) r1 = n;
  float s = 0.f, q = 0.f;
  for (int r = r0 + sub; r < r1; r += tpc) {
    float v = in[(size_t)r * D + col];
    s += v; q += v * v;
  }
  __shared__ float sh[512];
  sh[threadIdx.x] = s; sh[256 + threadIdx.x] = q;
  __syncthreads();
  if (sub == 0) {
    for (int k = 1; k < tpc; k++) { s += sh[col + k * D]; q += sh[256 + col + k * D]; }
    atomicAdd(&acc[col], s);
    atomicAdd(&acc[D + col], q);
  }
}

// ---------------- BN finalize: scale = g*rsqrt(var+eps), shift = b - mu*scale
__global__ void bn_finalize_k(const float* __restrict__ acc, const float* __restrict__ g,
                              const float* __restrict__ b, float* __restrict__ scale,
                              float* __restrict__ shift, int n, int d) {
  int c = threadIdx.x;
  if (c < d) {
    float inv_n = 1.f / (float)n;
    float mu  = acc[c] * inv_n;
    float var = acc[d + c] * inv_n - mu * mu;
    float sc  = g[c] * rsqrtf(var + BN_EPS);
    scale[c] = sc;
    shift[c] = b[c] - mu * sc;
  }
}

// ---------------- elementwise BN apply (+optional relu) ----------------
__global__ __launch_bounds__(256) void bn_apply_k(const float* __restrict__ in,
                                                  float* __restrict__ out,
                                                  const float* __restrict__ scale,
                                                  const float* __restrict__ shift,
                                                  int total, int dmask, int do_relu) {
  int i = blockIdx.x * 256 + threadIdx.x;
  if (i < total) {
    int c = i & dmask;
    float v = in[i] * scale[c] + shift[c];
    out[i] = do_relu ? fmaxf(v, 0.f) : v;
  }
}

// ---------------- edge phase: m = relu(h[src] + ea@lin_w^T + lin_b); scatter-add
// one wave (32 lanes) per edge; lanes cover IND columns (IND/32 each)
template <int IND>
__global__ __launch_bounds__(256) void edge_msg_k(const int* __restrict__ src,
                                                  const int* __restrict__ dst,
                                                  const float* __restrict__ ea,
                                                  const float* __restrict__ lw,
                                                  const float* __restrict__ lb,
                                                  const float* __restrict__ h,
                                                  float* __restrict__ agg) {
  int e = (blockIdx.x * 256 + threadIdx.x) >> 5;
  int lane = threadIdx.x & 31;
  if (e >= NE) return;
  int s = src[e], d = dst[e];
  float my = (lane < 16) ? ea[(size_t)e * 16 + lane] : 0.f;
  float ev[16];
#pragma unroll
  for (int k = 0; k < 16; k++) ev[k] = __shfl(my, k, 32);
#pragma unroll
  for (int j = 0; j < IND / 32; j++) {
    int col = lane + 32 * j;
    float a = lb[col];
#pragma unroll
    for (int k = 0; k < 16; k++) a = fmaf(ev[k], lw[col * 16 + k], a);
    float m = h[(size_t)s * IND + col] + a;
    m = fmaxf(m, 0.f);
    atomicAdd(&agg[(size_t)d * IND + col], m);
  }
}

// ---------------- WMMA 16x16 tile GEMM body (A from LDS f16, B = f16 weights) ---
// CDNA5 layouts (CDNA5 ISA): A 16x32 f16: lane row = lane&15, halves at
// K+off..K+off+7 and K+off+16..K+off+23, off = lane<16?0:8.
// B 32x16 f16: lane col = lane&15, halves at K+off2..K+off2+7 and K+off2+16..K+off2+23, off2 = lane<16?0:8.
// C/D f32: element (M = v + (lane>=16?8:0), N = lane&15) in c[v].
template <int K>
__device__ inline void gemm_tile(const _Float16* zs, const _Float16* __restrict__ wh,
                                 const float* __restrict__ bias, float* __restrict__ out,
                                 float* __restrict__ acc, int row0, int tid) {
  int lane  = tid & 31;
  int ntile = tid >> 5;            // 8 waves -> 8 column tiles (128 cols)
  int nrow  = lane & 15;
  int offA  = (lane < 16) ? 0 : 8;
  int offB  = (lane < 16) ? 0 : 8;
  int col   = ntile * 16 + nrow;
  v8f c = {};
#pragma unroll
  for (int kb = 0; kb < K; kb += 32) {
    const _Float16* ap = &zs[nrow * K + kb + offA];
    v8h a0 = *(const v8h*)ap;
    v8h a1 = *(const v8h*)(ap + 16);
    v16h a = __builtin_shufflevector(a0, a1, 0,1,2,3,4,5,6,7,8,9,10,11,12,13,14,15);
    const _Float16* bp = &wh[(size_t)col * K + kb + offB];
    v8h b0 = *(const v8h*)bp;
    v8h b1 = *(const v8h*)(bp + 16);
    v16h b = __builtin_shufflevector(b0, b1, 0,1,2,3,4,5,6,7,8,9,10,11,12,13,14,15);
    c = __builtin_amdgcn_wmma_f32_16x16x32_f16(false, a, false, b, (short)0, c,
                                               false, false);
  }
  float bv = bias[col];
  int mo = (lane < 16) ? 0 : 8;
  float s = 0.f, q = 0.f;
#pragma unroll
  for (int v = 0; v < 8; v++) {
    float val = c[v] + bv;
    out[(size_t)(row0 + mo + v) * 128 + col] = val;
    s += val; q += val * val;
  }
  // lanes l and l+16 hold the same column (rows 0-7 / 8-15): combine, then add
  s += __shfl_xor(s, 16, 32);
  q += __shfl_xor(q, 16, 32);
  if (lane < 16) {
    atomicAdd(&acc[col], s);
    atomicAdd(&acc[128 + col], q);
  }
}

// lin1: z = h + agg (f32->f16 tile in LDS); t = z @ w1^T + b1; fused BN stats
template <int K>
__global__ __launch_bounds__(256) void lin1_wmma_k(const float* __restrict__ h,
                                                   const float* __restrict__ agg,
                                                   const _Float16* __restrict__ wh,
                                                   const float* __restrict__ bias,
                                                   float* __restrict__ t,
                                                   float* __restrict__ acc) {
  __shared__ _Float16 zs[16 * K];
  int row0 = blockIdx.x * 16;
  size_t base = (size_t)row0 * K;
  for (int i = threadIdx.x; i < 16 * K; i += 256)
    zs[i] = (_Float16)(h[base + i] + agg[base + i]);
  __syncthreads();
  gemm_tile<K>(zs, wh, bias, t, acc, row0, threadIdx.x);
}

// lin2: u = relu(bn_in(t)) (f16 tile in LDS); y = u @ w2^T + b2; fused BN stats
__global__ __launch_bounds__(256) void lin2_wmma_k(const float* __restrict__ t,
                                                   const float* __restrict__ scale,
                                                   const float* __restrict__ shift,
                                                   const _Float16* __restrict__ wh,
                                                   const float* __restrict__ bias,
                                                   float* __restrict__ y,
                                                   float* __restrict__ acc) {
  __shared__ _Float16 us[16 * 128];
  int row0 = blockIdx.x * 16;
  size_t base = (size_t)row0 * 128;
  for (int i = threadIdx.x; i < 16 * 128; i += 256) {
    int cc = i & 127;
    float v = t[base + i] * scale[cc] + shift[cc];
    us[i] = (_Float16)fmaxf(v, 0.f);
  }
  __syncthreads();
  gemm_tile<128>(us, wh, bias, y, acc, row0, threadIdx.x);
}

// ---------------- pooling ----------------
__global__ __launch_bounds__(256) void pool_init_k(float* gsum, int* gmax, float* gcnt) {
  int i = blockIdx.x * 256 + threadIdx.x;
  if (i < NG * 128) { gsum[i] = 0.f; gmax[i] = (int)0xFF7FFFFFu; /* -FLT_MAX bits */ }
  if (i < NG) gcnt[i] = 0.f;
}

__global__ __launch_bounds__(256) void pool_acc_k(const float* __restrict__ h,
                                                  const int* __restrict__ batch,
                                                  float* __restrict__ gsum,
                                                  int* __restrict__ gmax,
                                                  float* __restrict__ gcnt) {
  int node = (blockIdx.x * 256 + threadIdx.x) >> 5;
  int lane = threadIdx.x & 31;
  if (node >= NN) return;
  int g = batch[node];
  if (lane == 0) atomicAdd(&gcnt[g], 1.f);
#pragma unroll
  for (int j = 0; j < 4; j++) {
    int c = lane + 32 * j;
    float v = h[(size_t)node * 128 + c];
    atomicAdd(&gsum[g * 128 + c], v);
    if (v >= 0.f) atomicMax(&gmax[g * 128 + c], __float_as_int(v));
    else          atomicMin((unsigned int*)&gmax[g * 128 + c], __float_as_uint(v));
  }
}

__global__ __launch_bounds__(256) void pool_fin_k(const float* __restrict__ gsum,
                                                  const int* __restrict__ gmax,
                                                  const float* __restrict__ gcnt,
                                                  float* __restrict__ out) {
  int i = blockIdx.x * 256 + threadIdx.x;
  if (i >= NG * 128) return;
  int g = i >> 7, c = i & 127;
  float cnt = gcnt[g];
  float mean = gsum[i] / fmaxf(cnt, 1.f);
  float mx = (cnt > 0.f) ? __int_as_float(gmax[i]) : 0.f;
  out[(size_t)g * 256 + c] = mean;
  out[(size_t)g * 256 + 128 + c] = mx;
}

// ---------------- host orchestration ----------------
extern "C" void kernel_launch(void* const* d_in, const int* in_sizes, int n_in,
                              void* d_out, int out_size, void* d_ws, size_t ws_size,
                              hipStream_t stream) {
  (void)in_sizes; (void)n_in; (void)out_size; (void)ws_size;
  const float* x     = (const float*)d_in[0];
  const int*   ei    = (const int*)d_in[1];
  const float* ea    = (const float*)d_in[2];
  const int*   batch = (const int*)d_in[3];
  const float* bn0g  = (const float*)d_in[4];
  const float* bn0b  = (const float*)d_in[5];
  const int* src = ei;
  const int* dst = ei + NE;

  // workspace carve
  char* p = (char*)d_ws;
  auto carve = [&](size_t bytes) { char* r = p; p += (bytes + 255) & ~(size_t)255; return r; };
  float* hbuf = (float*)carve((size_t)NN * 128 * 4);
  float* agg  = (float*)carve((size_t)NN * 128 * 4);
  float* tbuf = (float*)carve((size_t)NN * 128 * 4);
  float* bnws = (float*)carve(1024 * 4);
  float* acc1 = bnws;        float* scale1 = bnws + 256; float* shift1 = bnws + 384;
  float* acc2 = bnws + 512;  float* scale2 = bnws + 768; float* shift2 = bnws + 896;
  float* gsum = (float*)carve((size_t)NG * 128 * 4);
  int*   gmax = (int*)carve((size_t)NG * 128 * 4);
  float* gcnt = (float*)carve((size_t)NG * 4);
  _Float16* w1h[3];
  _Float16* w2h[3];
  w1h[0] = (_Float16*)carve(128 * 64 * 2);
  w2h[0] = (_Float16*)carve(128 * 128 * 2);
  w1h[1] = (_Float16*)carve(128 * 128 * 2);
  w2h[1] = (_Float16*)carve(128 * 128 * 2);
  w1h[2] = (_Float16*)carve(128 * 128 * 2);
  w2h[2] = (_Float16*)carve(128 * 128 * 2);

  // convert node-MLP weights to f16 (input indices: w1 at base+2, w2 at base+6)
  const int bases[3] = {6, 16, 26};
  const int w1n[3] = {128 * 64, 128 * 128, 128 * 128};
  for (int l = 0; l < 3; l++) {
    cvt_f16_k<<<(w1n[l] + 255) / 256, 256, 0, stream>>>(
        (const float*)d_in[bases[l] + 2], w1h[l], w1n[l]);
    cvt_f16_k<<<(128 * 128 + 255) / 256, 256, 0, stream>>>(
        (const float*)d_in[bases[l] + 6], w2h[l], 128 * 128);
  }

  // ---- bn0: h0 = bn(x)  [N,64] ----
  hipMemsetAsync(bnws, 0, 1024 * 4, stream);
  col_stats_k<64><<<(NN + 255) / 256, 256, 0, stream>>>(x, acc1, NN, 256);
  bn_finalize_k<<<1, 128, 0, stream>>>(acc1, bn0g, bn0b, scale1, shift1, NN, 64);
  bn_apply_k<<<(NN * 64 + 255) / 256, 256, 0, stream>>>(x, hbuf, scale1, shift1,
                                                        NN * 64, 63, 0);

  // ---- conv layers ----
  for (int l = 0; l < 3; l++) {
    int base = bases[l];
    int ind = (l == 0) ? 64 : 128;
    int relu_out = (l == 2) ? 0 : 1;
    const float* lw   = (const float*)d_in[base + 0];
    const float* lb   = (const float*)d_in[base + 1];
    const float* b1   = (const float*)d_in[base + 3];
    const float* bng  = (const float*)d_in[base + 4];
    const float* bnb  = (const float*)d_in[base + 5];
    const float* b2   = (const float*)d_in[base + 7];
    const float* obng = (const float*)d_in[base + 8];  // bn1/bn2/bn3 gamma
    const float* obnb = (const float*)d_in[base + 9];

    hipMemsetAsync(agg, 0, (size_t)NN * ind * 4, stream);
    hipMemsetAsync(bnws, 0, 1024 * 4, stream);

    if (ind == 64)
      edge_msg_k<64><<<NE / 8, 256, 0, stream>>>(src, dst, ea, lw, lb, hbuf, agg);
    else
      edge_msg_k<128><<<NE / 8, 256, 0, stream>>>(src, dst, ea, lw, lb, hbuf, agg);

    if (ind == 64)
      lin1_wmma_k<64><<<NN / 16, 256, 0, stream>>>(hbuf, agg, w1h[l], b1, tbuf, acc1);
    else
      lin1_wmma_k<128><<<NN / 16, 256, 0, stream>>>(hbuf, agg, w1h[l], b1, tbuf, acc1);

    bn_finalize_k<<<1, 128, 0, stream>>>(acc1, bng, bnb, scale1, shift1, NN, 128);
    // y written into agg (free after lin1)
    lin2_wmma_k<<<NN / 16, 256, 0, stream>>>(tbuf, scale1, shift1, w2h[l], b2, agg, acc2);
    bn_finalize_k<<<1, 128, 0, stream>>>(acc2, obng, obnb, scale2, shift2, NN, 128);
    bn_apply_k<<<(NN * 128 + 255) / 256, 256, 0, stream>>>(agg, hbuf, scale2, shift2,
                                                           NN * 128, 127, relu_out);
  }

  // ---- pooling: [G, mean||max] ----
  pool_init_k<<<(NG * 128 + 255) / 256, 256, 0, stream>>>(gsum, gmax, gcnt);
  pool_acc_k<<<NN / 8, 256, 0, stream>>>(hbuf, batch, gsum, gmax, gcnt);
  pool_fin_k<<<(NG * 128 + 255) / 256, 256, 0, stream>>>(gsum, gmax, gcnt,
                                                         (float*)d_out);
}
